// Block_82325933129820
// MI455X (gfx1250) — matrix operand order repaired and checked by reading, not compiled, not run on hardware
//
#include <hip/hip_runtime.h>

// ---------------------------------------------------------------------------
// Types
// ---------------------------------------------------------------------------
typedef __attribute__((ext_vector_type(8)))  float   v8f;
typedef __attribute__((ext_vector_type(8)))  __bf16  v8bf;
typedef __attribute__((ext_vector_type(16))) __bf16  v16bf;

#define NB   4
#define TT   1024
#define EE   1024
#define QKVD 2048
#define NH   16
#define DH   128
#define BT   4096          // NB*TT
#define EPSF 1.1920929e-7f

__device__ __forceinline__ __bf16 f2bf(float f) {
    unsigned u = __builtin_bit_cast(unsigned, f);
    unsigned short r = (unsigned short)((u + 0x7FFFu + ((u >> 16) & 1u)) >> 16);
    return __builtin_bit_cast(__bf16, r);
}

__device__ __forceinline__ v8f wmma_bf16(v16bf a, v16bf b, v8f c) {
    return __builtin_amdgcn_wmma_f32_16x16x32_bf16(false, a, false, b, (short)0, c,
                                                   false, false);
}

__device__ __forceinline__ v8f zero8() {
    v8f z = {0.f, 0.f, 0.f, 0.f, 0.f, 0.f, 0.f, 0.f};
    return z;
}

// ---------------------------------------------------------------------------
// fp32 -> bf16 elementwise conversion (weights)
// ---------------------------------------------------------------------------
__global__ __launch_bounds__(256) void cvt_bf16(const float* __restrict__ X,
                                                __bf16* __restrict__ O, int n) {
    int i = blockIdx.x * 256 + threadIdx.x;
    if (i < n) O[i] = f2bf(X[i]);
}

// ---------------------------------------------------------------------------
// Row RMS-normalize + cast to bf16 (one 256-thread block per row)
// ---------------------------------------------------------------------------
__global__ __launch_bounds__(256) void rms_cast(const float* __restrict__ X,
                                                __bf16* __restrict__ O, int L) {
    int row = blockIdx.x;
    const float* xr = X + (size_t)row * L;
    float ss = 0.f;
    for (int i = threadIdx.x; i < L; i += 256) { float v = xr[i]; ss += v * v; }
#pragma unroll
    for (int off = 16; off; off >>= 1) ss += __shfl_xor(ss, off);
    __shared__ float red[8];
    if ((threadIdx.x & 31) == 0) red[threadIdx.x >> 5] = ss;
    __syncthreads();
    float tot = 0.f;
#pragma unroll
    for (int w = 0; w < 8; w++) tot += red[w];
    float s = rsqrtf(tot / (float)L + EPSF);
    __bf16* orow = O + (size_t)row * L;
    for (int i = threadIdx.x; i < L; i += 256) orow[i] = f2bf(xr[i] * s);
}

// ---------------------------------------------------------------------------
// Row RMS scale only (strided rows inside qkv buffer)
// ---------------------------------------------------------------------------
__global__ __launch_bounds__(256) void rms_scale(const float* __restrict__ X,
                                                 float* __restrict__ S, int L,
                                                 int stride, int off0) {
    int row = blockIdx.x;
    const float* xr = X + (size_t)row * stride + off0;
    float ss = 0.f;
    for (int i = threadIdx.x; i < L; i += 256) { float v = xr[i]; ss += v * v; }
#pragma unroll
    for (int off = 16; off; off >>= 1) ss += __shfl_xor(ss, off);
    __shared__ float red[8];
    if ((threadIdx.x & 31) == 0) red[threadIdx.x >> 5] = ss;
    __syncthreads();
    if (threadIdx.x == 0) {
        float tot = 0.f;
#pragma unroll
        for (int w = 0; w < 8; w++) tot += red[w];
        S[row] = rsqrtf(tot / (float)L + EPSF);
    }
}

// ---------------------------------------------------------------------------
// bf16 GEMM: C[M,N] = A[M,K] @ W[N,K]^T (+ optional residual, compile-time).
// 8 waves/block as 4(M) x 2(N); wave tile 32x64 (2x4 wmma frags);
// block tile 128x128. f32 accumulate, bf16 inputs via v_wmma_f32_16x16x32_bf16.
// ---------------------------------------------------------------------------
template <bool HASR>
__global__ __launch_bounds__(256) void gemm_bf16(const __bf16* __restrict__ A,
                                                 const __bf16* __restrict__ W,
                                                 const float* __restrict__ R,
                                                 float* __restrict__ C,
                                                 int M, int N, int K) {
    const int lane = threadIdx.x & 31;
    const int w    = threadIdx.x >> 5;
    const int rlo  = lane & 15;
    const int hi   = lane >> 4;
    const int mBase = blockIdx.y * 128 + (w >> 1) * 32;   // 4 m-waves
    const int nBase = blockIdx.x * 128 + (w & 1) * 64;    // 2 n-waves

    v8f acc[2][4];
#pragma unroll
    for (int mt = 0; mt < 2; mt++)
#pragma unroll
        for (int nt = 0; nt < 4; nt++) acc[mt][nt] = zero8();

    const __bf16* arow[2];
    const __bf16* wrow[4];
#pragma unroll
    for (int mt = 0; mt < 2; mt++)
        arow[mt] = A + (size_t)(mBase + mt * 16 + rlo) * K;
#pragma unroll
    for (int nt = 0; nt < 4; nt++)
        wrow[nt] = W + (size_t)(nBase + nt * 16 + rlo) * K;

    for (int kk = 0; kk < K; kk += 32) {
        // prefetch the k-block 512 elements (16 blocks) ahead into cache (global_prefetch_b8)
        if (kk + 544 <= K) {
#pragma unroll
            for (int mt = 0; mt < 2; mt++)
                __builtin_prefetch(arow[mt] + kk + 512, 0, 1);
#pragma unroll
            for (int nt = 0; nt < 4; nt++)
                __builtin_prefetch(wrow[nt] + kk + 512, 0, 1);
        }
        v16bf af[2], bfm[4];
#pragma unroll
        for (int mt = 0; mt < 2; mt++) {
            int kb = kk + hi * 16;
            v8bf lo = *(const v8bf*)(arow[mt] + kb);
            v8bf h8 = *(const v8bf*)(arow[mt] + kb + 8);
#pragma unroll
            for (int i = 0; i < 8; i++) { af[mt][i] = lo[i]; af[mt][i + 8] = h8[i]; }
        }
#pragma unroll
        for (int nt = 0; nt < 4; nt++)
            bfm[nt] = *(const v16bf*)(wrow[nt] + kk + hi * 16);
#pragma unroll
        for (int mt = 0; mt < 2; mt++)
#pragma unroll
            for (int nt = 0; nt < 4; nt++)
                acc[mt][nt] = wmma_bf16(af[mt], bfm[nt], acc[mt][nt]);
    }

#pragma unroll
    for (int mt = 0; mt < 2; mt++)
#pragma unroll
        for (int nt = 0; nt < 4; nt++)
#pragma unroll
            for (int j = 0; j < 8; j++) {
                int m = mBase + mt * 16 + j + hi * 8;
                int n = nBase + nt * 16 + rlo;
                size_t o = (size_t)m * N + n;
                float v = acc[mt][nt][j];
                if (HASR) v += R[o];
                C[o] = v;
            }
}

// ---------------------------------------------------------------------------
// AFT cumulative scan over T per (batch, channel)
// ---------------------------------------------------------------------------
__global__ __launch_bounds__(256) void aft_scan(const float* __restrict__ qkv,
                                                const float* __restrict__ sq,
                                                const float* __restrict__ sk,
                                                __bf16* __restrict__ Y) {
    int idx = blockIdx.x * 256 + threadIdx.x;     // b*QKVD + c
    if (idx >= NB * QKVD) return;
    int b = idx >> 11, c = idx & (QKVD - 1);
    float accw = 0.f, accwv = 0.f;
    for (int t = 0; t < TT; t++) {
        int bt = b * TT + t;
        size_t base = (size_t)bt * (3 * QKVD);
        float q = qkv[base + c] * sq[bt];
        float k = qkv[base + QKVD + c] * sk[bt];
        float v = qkv[base + 2 * QKVD + c];
        float wv = __expf(k);
        accw += wv;
        accwv += wv * v;
        float y = (1.f / (1.f + __expf(-q))) * (accwv / (accw + 1e-6f));
        Y[(size_t)bt * QKVD + c] = f2bf(y);
    }
}

// ---------------------------------------------------------------------------
// RoPE + per-head RMS, emit bf16 Q/K (bh,T,DH) and transposed V (bh,DH,T)
// ---------------------------------------------------------------------------
__global__ __launch_bounds__(128) void rope_rms(const float* __restrict__ qkv,
                                                const float* __restrict__ cosT,
                                                const float* __restrict__ sinT,
                                                __bf16* __restrict__ Qo,
                                                __bf16* __restrict__ Ko,
                                                __bf16* __restrict__ Vt) {
    int bt = blockIdx.x, h = blockIdx.y, d = threadIdx.x;
    int t = bt & (TT - 1);
    const float* base = qkv + (size_t)bt * (3 * QKVD) + h * (3 * DH);
    int dr = d & 63;
    float c = cosT[t * 64 + dr], s = sinT[t * 64 + dr];
    float qr, kr;
    if (d < 64) {
        qr = base[d] * c + base[d + 64] * s;
        kr = base[128 + d] * c + base[192 + d] * s;
    } else {
        qr = -base[d - 64] * s + base[d] * c;
        kr = -base[64 + d] * s + base[128 + d] * c;
    }
    float v = base[256 + d];
    float q2 = qr * qr, k2 = kr * kr;
#pragma unroll
    for (int off = 16; off; off >>= 1) {
        q2 += __shfl_xor(q2, off);
        k2 += __shfl_xor(k2, off);
    }
    __shared__ float rq[4], rk[4];
    int wid = threadIdx.x >> 5;
    if ((threadIdx.x & 31) == 0) { rq[wid] = q2; rk[wid] = k2; }
    __syncthreads();
    float sumq = rq[0] + rq[1] + rq[2] + rq[3];
    float sumk = rk[0] + rk[1] + rk[2] + rk[3];
    float qs  = rsqrtf(sumq / 128.f + EPSF);
    float ksc = rsqrtf(sumk / 128.f + EPSF);
    int bh = (bt >> 10) * NH + h;
    size_t ro = ((size_t)bh * TT + t) * DH + d;
    Qo[ro] = f2bf(qr * qs);
    Ko[ro] = f2bf(kr * ksc);
    Vt[((size_t)bh * DH + d) * TT + t] = f2bf(v);
}

// ---------------------------------------------------------------------------
// Flash attention: one wave per (bh, 16-query tile). Online softmax,
// S = Q.K^T and O += P.V through WMMA; P transposed through LDS.
// ---------------------------------------------------------------------------
__global__ __launch_bounds__(32) void flash_attn(const __bf16* __restrict__ Q,
                                                 const __bf16* __restrict__ Kt,
                                                 const __bf16* __restrict__ Vt,
                                                 __bf16* __restrict__ Y) {
    __shared__ float sp[16 * 32];   // [key c][row m] col-major: sp[c*16+m]
    const int lane = threadIdx.x & 31;
    const int rlo = lane & 15;
    const int hi  = lane >> 4;
    const int qt = blockIdx.x;
    const int bh = blockIdx.y;
    const int qstart = qt * 16;
    const float scale = 0.08838834764831845f;   // 1/sqrt(128)

    // Q fragments: 4 chunks of K=32 over head dim
    v16bf qf[4];
    {
        const __bf16* qbase = Q + ((size_t)bh * TT + qstart + rlo) * DH;
#pragma unroll
        for (int kc = 0; kc < 4; kc++) {
            int kb = kc * 32 + hi * 16;
            v8bf lo = *(const v8bf*)(qbase + kb);
            v8bf h8 = *(const v8bf*)(qbase + kb + 8);
#pragma unroll
            for (int i = 0; i < 8; i++) { qf[kc][i] = lo[i]; qf[kc][i + 8] = h8[i]; }
        }
    }

    float rowmax[8], rowsum[8];
    v8f acc[8];
#pragma unroll
    for (int j = 0; j < 8; j++) { rowmax[j] = -3.0e38f; rowsum[j] = 0.f; }
#pragma unroll
    for (int nt = 0; nt < 8; nt++) acc[nt] = zero8();

    for (int ks = 0; ks <= qstart; ks += 32) {
        v8f s0 = zero8(), s1 = zero8();
        const __bf16* kb0 = Kt + ((size_t)bh * TT + ks + rlo) * DH + hi * 16;
        const __bf16* kb1 = kb0 + (size_t)16 * DH;
#pragma unroll
        for (int kc = 0; kc < 4; kc++) {
            v16bf b0 = *(const v16bf*)(kb0 + kc * 32);
            s0 = wmma_bf16(qf[kc], b0, s0);
            v16bf b1 = *(const v16bf*)(kb1 + kc * 32);
            s1 = wmma_bf16(qf[kc], b1, s1);
        }
        s0 = s0 * scale;
        s1 = s1 * scale;
        // causal mask
#pragma unroll
        for (int j = 0; j < 8; j++) {
            int m = qstart + j + hi * 8;
            if (ks + rlo > m)      s0[j] = -3.0e38f;
            if (ks + 16 + rlo > m) s1[j] = -3.0e38f;
        }
        // online softmax
        float nm[8], corr[8];
#pragma unroll
        for (int j = 0; j < 8; j++) {
            float cm = fmaxf(s0[j], s1[j]);
#pragma unroll
            for (int off = 1; off < 16; off <<= 1) cm = fmaxf(cm, __shfl_xor(cm, off));
            nm[j] = fmaxf(rowmax[j], cm);
            corr[j] = __expf(rowmax[j] - nm[j]);
            rowmax[j] = nm[j];
        }
        v8f p0, p1;
#pragma unroll
        for (int j = 0; j < 8; j++) {
            p0[j] = __expf(s0[j] - nm[j]);
            p1[j] = __expf(s1[j] - nm[j]);
            float l = p0[j] + p1[j];
#pragma unroll
            for (int off = 1; off < 16; off <<= 1) l += __shfl_xor(l, off);
            rowsum[j] = rowsum[j] * corr[j] + l;
        }
#pragma unroll
        for (int nt = 0; nt < 8; nt++)
#pragma unroll
            for (int j = 0; j < 8; j++) acc[nt][j] *= corr[j];

        // transpose P through LDS into WMMA A-fragment layout
#pragma unroll
        for (int j = 0; j < 8; j++) {
            sp[rlo * 16 + hi * 8 + j]        = p0[j];
            sp[(16 + rlo) * 16 + hi * 8 + j] = p1[j];
        }
        __syncthreads();
        v16bf pf;
#pragma unroll
        for (int i = 0; i < 16; i++) {
            int ki = hi * 16 + i;
            pf[i] = f2bf(sp[ki * 16 + rlo]);
        }
        __syncthreads();

        // O += P.V  (V pre-transposed: (bh, DH, T))
        const __bf16* vbase = Vt + (size_t)bh * DH * TT + ks + hi * 16;
#pragma unroll
        for (int nt = 0; nt < 8; nt++) {
            v16bf vf = *(const v16bf*)(vbase + (size_t)(nt * 16 + rlo) * TT);
            acc[nt] = wmma_bf16(pf, vf, acc[nt]);
        }
    }

    int b = bh >> 4, h = bh & 15;
#pragma unroll
    for (int nt = 0; nt < 8; nt++)
#pragma unroll
        for (int j = 0; j < 8; j++) {
            int m = qstart + j + hi * 8;
            float val = acc[nt][j] / rowsum[j];
            Y[((size_t)(b * TT + m)) * QKVD + h * DH + nt * 16 + rlo] = f2bf(val);
        }
}

// ---------------------------------------------------------------------------
// SwiGLU: h = u * silu(g)
// ---------------------------------------------------------------------------
__global__ __launch_bounds__(256) void swiglu(const float* __restrict__ UG,
                                              __bf16* __restrict__ Hb, int n) {
    int i = blockIdx.x * 256 + threadIdx.x;
    if (i >= n) return;
    int r = i >> 10, c = i & (EE - 1);
    float u = UG[(size_t)r * (2 * EE) + c];
    float g = UG[(size_t)r * (2 * EE) + EE + c];
    float sg = 1.f / (1.f + __expf(-g));
    Hb[i] = f2bf(u * g * sg);
}

// ---------------------------------------------------------------------------
// Host launcher
// ---------------------------------------------------------------------------
static char* carve(char*& p, size_t bytes) {
    char* r = p;
    p += (bytes + 255) & ~(size_t)255;
    return r;
}

extern "C" void kernel_launch(void* const* d_in, const int* in_sizes, int n_in,
                              void* d_out, int out_size, void* d_ws, size_t ws_size,
                              hipStream_t stream) {
    (void)in_sizes; (void)n_in; (void)out_size; (void)ws_size;

    const float* x    = (const float*)d_in[0];
    const float* cosp = (const float*)d_in[1];
    const float* sinp = (const float*)d_in[2];

    char* p = (char*)d_ws;
    __bf16* wq_bf  = (__bf16*)carve(p, (size_t)3 * QKVD * EE * 2);
    __bf16* wsw_bf = (__bf16*)carve(p, (size_t)2 * EE * QKVD * 2);
    __bf16* wo_bf  = (__bf16*)carve(p, (size_t)EE * EE * 2);
    __bf16* xn_bf  = (__bf16*)carve(p, (size_t)BT * EE * 2);
    float*  qkvf   = (float*)carve(p, (size_t)BT * 3 * QKVD * 4);
    float*  sq     = (float*)carve(p, (size_t)BT * 4);
    float*  sk     = (float*)carve(p, (size_t)BT * 4);
    __bf16* y_bf   = (__bf16*)carve(p, (size_t)BT * QKVD * 2);
    float*  ugf    = (float*)carve(p, (size_t)BT * 2 * EE * 4);
    __bf16* h_bf   = (__bf16*)carve(p, (size_t)BT * EE * 2);
    float*  xb0    = (float*)carve(p, (size_t)BT * EE * 4);
    float*  xb1    = (float*)carve(p, (size_t)BT * EE * 4);
    __bf16* q_bf   = (__bf16*)carve(p, (size_t)NB * NH * TT * DH * 2 + 32768);
    __bf16* k_bf   = (__bf16*)carve(p, (size_t)NB * NH * TT * DH * 2 + 32768);
    __bf16* vT_bf  = (__bf16*)carve(p, (size_t)NB * NH * TT * DH * 2 + 32768);

    const float* xin = x;
    const int nqkv = 3 * QKVD * EE, nsw = 2 * EE * QKVD, nout = EE * EE;

    for (int l = 0; l < 4; l++) {
        const float* wqkv = (const float*)d_in[3 + 3 * l];
        const float* wsw  = (const float*)d_in[4 + 3 * l];
        const float* wout = (const float*)d_in[5 + 3 * l];

        cvt_bf16<<<(nqkv + 255) / 256, 256, 0, stream>>>(wqkv, wq_bf, nqkv);
        cvt_bf16<<<(nsw  + 255) / 256, 256, 0, stream>>>(wsw,  wsw_bf, nsw);
        cvt_bf16<<<(nout + 255) / 256, 256, 0, stream>>>(wout, wo_bf, nout);

        rms_cast<<<BT, 256, 0, stream>>>(xin, xn_bf, EE);
        gemm_bf16<false><<<dim3(3 * QKVD / 128, BT / 128), 256, 0, stream>>>(
            xn_bf, wq_bf, nullptr, qkvf, BT, 3 * QKVD, EE);

        if (l < 3) {
            rms_scale<<<BT, 256, 0, stream>>>(qkvf, sq, QKVD, 3 * QKVD, 0);
            rms_scale<<<BT, 256, 0, stream>>>(qkvf, sk, QKVD, 3 * QKVD, QKVD);
            aft_scan<<<(NB * QKVD) / 256, 256, 0, stream>>>(qkvf, sq, sk, y_bf);
        } else {
            rope_rms<<<dim3(BT, NH), 128, 0, stream>>>(qkvf, cosp, sinp,
                                                       q_bf, k_bf, vT_bf);
            flash_attn<<<dim3(TT / 16, NB * NH), 32, 0, stream>>>(
                q_bf, k_bf, vT_bf, y_bf);
        }

        gemm_bf16<false><<<dim3(2 * EE / 128, BT / 128), 256, 0, stream>>>(
            y_bf, wsw_bf, nullptr, ugf, BT, 2 * EE, QKVD);
        swiglu<<<(BT * EE) / 256, 256, 0, stream>>>(ugf, h_bf, BT * EE);

        float* xout = (l == 3) ? (float*)d_out : ((l & 1) ? xb1 : xb0);
        gemm_bf16<true><<<dim3(EE / 128, BT / 128), 256, 0, stream>>>(
            h_bf, wo_bf, xin, xout, BT, EE, EE);
        xin = xout;
    }
}
